// MMEncoder_5901285064863
// MI455X (gfx1250) — compile-verified, matrix operand order checked by reading, not run on hardware
//
#include <hip/hip_runtime.h>
#include <math.h>

typedef __bf16 bf16_t;
typedef __attribute__((ext_vector_type(16))) __bf16 v16bf;
typedef __attribute__((ext_vector_type(8)))  __bf16 v8bf;
typedef __attribute__((ext_vector_type(8)))  float  v8f;
typedef __attribute__((ext_vector_type(4)))  int    v4i;

#define DIMX   768
#define HEADS  12
#define HDIM   64
#define INNERX 768
#define DFFX   3072
#define SEQ    1024
#define BATCH  4
#define ROWS   (BATCH*SEQ)      // 4096
#define NLAYER 8
#define QKVW   (3*INNERX)       // 2304
#define SCALEF 0.125f           // 1/sqrt(64)

// ---------- CDNA5 async global->LDS path (guarded; falls back to sync) ----
#if defined(__AMDGCN__) && __has_builtin(__builtin_amdgcn_global_load_async_to_lds_b128) && \
    __has_builtin(__builtin_amdgcn_s_wait_asynccnt)
#define ASYNC_LDS 1
#else
#define ASYNC_LDS 0
#endif

#if ASYNC_LDS
#define WAIT_ASYNC(n) __builtin_amdgcn_s_wait_asynccnt(n)
#else
#define WAIT_ASYNC(n)
#endif

// Copy 16B per lane from global to LDS (async when available).
static __device__ __forceinline__ void cp16(const bf16_t* g, bf16_t* l) {
#if ASYNC_LDS
  __builtin_amdgcn_global_load_async_to_lds_b128(
      (__attribute__((address_space(1))) v4i*)g,
      (__attribute__((address_space(3))) v4i*)l, 0, 0);
#else
  *(v8bf*)l = *(const v8bf*)g;
#endif
}

// ---------- helpers ----------
static __device__ __forceinline__ v8f zero8() {
  v8f z;
#pragma unroll
  for (int i = 0; i < 8; ++i) z[i] = 0.f;
  return z;
}

// Assemble a 16x32 bf16 A/B fragment row for one lane from LDS.
// rowbase points at element [row][chunk_k0]; ISA layout: lane half picks
// K = half*8+0..7 and K = 16+half*8+0..7 (two contiguous 16B chunks).
static __device__ __forceinline__ v16bf load_frag(const bf16_t* rowbase, int half) {
  v8bf lo = *(const v8bf*)(rowbase + half * 8);
  v8bf hi = *(const v8bf*)(rowbase + 16 + half * 8);
  v16bf r;
#pragma unroll
  for (int i = 0; i < 8; ++i) { r[i] = lo[i]; r[i + 8] = hi[i]; }
  return r;
}

static __device__ __forceinline__ v8f wmma_bf16(v16bf a, v16bf b, v8f c) {
  return __builtin_amdgcn_wmma_f32_16x16x32_bf16(false, a, false, b, (short)0, c,
                                                 false, false);
}

// Tiled f32[K x N] -> bf16 transposed [N x K]; grid.z = layer, block (32,8).
__global__ __launch_bounds__(256) void cvtT_kernel(const float* __restrict__ in,
                                                   bf16_t* __restrict__ out,
                                                   int K, int N) {
  __shared__ float tile[32][33];
  const size_t lbase = (size_t)blockIdx.z * K * N;
  const float* inl = in + lbase;
  bf16_t* outl = out + lbase;
  const int n0 = blockIdx.x * 32, k0 = blockIdx.y * 32;
  const int tx = threadIdx.x, ty = threadIdx.y;
#pragma unroll
  for (int i = 0; i < 4; ++i)
    tile[ty + i * 8][tx] = inl[(size_t)(k0 + ty + i * 8) * N + n0 + tx];
  __syncthreads();
#pragma unroll
  for (int i = 0; i < 4; ++i)
    outl[(size_t)(n0 + ty + i * 8) * K + k0 + tx] = (bf16_t)tile[tx][ty + i * 8];
}

// Transpose V out of the fused qkv activation once per layer:
// vT[(b*HEADS+h)*HDIM + d][n] = qkv[(b*SEQ+n)*QKVW + 2*INNER + h*HDIM + d]
__global__ __launch_bounds__(256) void transposeV_kernel(
    const bf16_t* __restrict__ qkv, bf16_t* __restrict__ vT) {
  __shared__ bf16_t tile[32][33];
  const int bh = blockIdx.z;
  const int b = bh / HEADS, h = bh % HEADS;
  const int n0 = blockIdx.x * 32, d0 = blockIdx.y * 32;
  const int tx = threadIdx.x, ty = threadIdx.y;
  const size_t vbase = 2 * INNERX + (size_t)h * HDIM;
#pragma unroll
  for (int i = 0; i < 4; ++i)
    tile[ty + i * 8][tx] =
        qkv[(size_t)(b * SEQ + n0 + ty + i * 8) * QKVW + vbase + d0 + tx];
  __syncthreads();
#pragma unroll
  for (int i = 0; i < 4; ++i)
    vT[((size_t)bh * HDIM + d0 + ty + i * 8) * SEQ + n0 + tx] =
        tile[tx][ty + i * 8];
}

__global__ void copy_f32_kernel(const float* __restrict__ in, float* __restrict__ out, int n) {
  int i = blockIdx.x * blockDim.x + threadIdx.x;
  int stride = gridDim.x * blockDim.x;
  for (; i < n; i += stride) out[i] = in[i];
}

// ---------- LayerNorm (768 cols, 256 threads, 3 elems/thread) ----------
template <bool BF>
__global__ __launch_bounds__(256) void layernorm_kernel(
    const float* __restrict__ x, const float* __restrict__ g, const float* __restrict__ bta,
    bf16_t* __restrict__ outB, float* __restrict__ outF) {
  const int row = blockIdx.x;
  const int tid = threadIdx.x;
  const float* xr = x + (size_t)row * DIMX;
  float v[3];
  float s1 = 0.f, s2 = 0.f;
#pragma unroll
  for (int i = 0; i < 3; ++i) {
    v[i] = xr[tid + i * 256];
    s1 += v[i];
    s2 += v[i] * v[i];
  }
#pragma unroll
  for (int off = 16; off >= 1; off >>= 1) {
    s1 += __shfl_xor(s1, off, 32);
    s2 += __shfl_xor(s2, off, 32);
  }
  __shared__ float r1[8], r2[8];
  if ((tid & 31) == 0) { r1[tid >> 5] = s1; r2[tid >> 5] = s2; }
  __syncthreads();
  float t1 = 0.f, t2 = 0.f;
#pragma unroll
  for (int i = 0; i < 8; ++i) { t1 += r1[i]; t2 += r2[i]; }
  const float mean = t1 * (1.f / DIMX);
  const float var = t2 * (1.f / DIMX) - mean * mean;
  const float rstd = rsqrtf(var + 1e-5f);
#pragma unroll
  for (int i = 0; i < 3; ++i) {
    int c = tid + i * 256;
    float y = (v[i] - mean) * rstd * g[c] + bta[c];
    if (BF) outB[(size_t)row * DIMX + c] = (bf16_t)y;
    else    outF[(size_t)row * DIMX + c] = y;
  }
}

// ---------- bf16 GEMM: C[MxN] = A[MxK] * Bt[NxK]^T (+bias, epilogue) -------
// Bt is the PRE-TRANSPOSED weight (N x K row-major) so both LDS tiles stage
// as contiguous 16B chunks (async global->LDS, double buffered).
// Block: 128 thr = 4 waves; tile 64(M) x 128(N); wave owns 64x32 strip.
// EPI: 1 = store bf16, 2 = exact-GELU then bf16, 3 = residual add, store f32
template <int EPI>
__global__ __launch_bounds__(128) void gemm_bf16_kernel(
    const bf16_t* __restrict__ A, const bf16_t* __restrict__ Bt,
    const float* __restrict__ bias, const float* __restrict__ resid,
    float* __restrict__ outF, bf16_t* __restrict__ outB, int M, int N, int K) {
  __shared__ bf16_t As[2][64][48];     // [buf][m][k], stride 48 keeps 16B align
  __shared__ bf16_t Bts[2][128][48];   // [buf][n][k]

  const int tid = threadIdx.x;
  const int lane = tid & 31;
  const int wave = tid >> 5;           // n-strip: wave*32
  const int half = lane >> 4;
  const int l15 = lane & 15;
  const int bm = blockIdx.y, bn = blockIdx.x;

  v8f acc[4][2];
#pragma unroll
  for (int i = 0; i < 4; ++i)
#pragma unroll
    for (int j = 0; j < 2; ++j) acc[i][j] = zero8();

  const int nIter = K / 32;
  auto stage = [&](int buf, int k0) {
#pragma unroll
    for (int i = 0; i < 2; ++i) {      // A: 64x32 = 256 chunks
      int u = tid + i * 128;
      int r = u >> 2, c8 = (u & 3) * 8;
      cp16(A + (size_t)(bm * 64 + r) * K + k0 + c8, &As[buf][r][c8]);
    }
#pragma unroll
    for (int i = 0; i < 4; ++i) {      // B: 128x32 = 512 chunks
      int u = tid + i * 128;
      int n = u >> 2, c8 = (u & 3) * 8;
      cp16(Bt + (size_t)(bn * 128 + n) * K + k0 + c8, &Bts[buf][n][c8]);
    }
  };

  stage(0, 0);
  for (int it = 0; it < nIter; ++it) {
    const int buf = it & 1;
    if (it + 1 < nIter) {
      stage(buf ^ 1, (it + 1) * 32);   // prefetch next tile into other buffer
      WAIT_ASYNC(6);                   // wait for current tile's 6 copies
    } else {
      WAIT_ASYNC(0);
    }
    __syncthreads();

    v16bf bfrag[2];
#pragma unroll
    for (int tn = 0; tn < 2; ++tn)
      bfrag[tn] = load_frag(&Bts[buf][wave * 32 + tn * 16 + l15][0], half);
#pragma unroll
    for (int tm = 0; tm < 4; ++tm) {
      v16bf afrag = load_frag(&As[buf][tm * 16 + l15][0], half);
#pragma unroll
      for (int tn = 0; tn < 2; ++tn)
        acc[tm][tn] = wmma_bf16(afrag, bfrag[tn], acc[tm][tn]);
    }
    __syncthreads();
  }

  // Epilogue. D layout: VGPR r, lanes0-15 -> M=r, lanes16-31 -> M=r+8; N=l15.
#pragma unroll
  for (int tm = 0; tm < 4; ++tm) {
#pragma unroll
    for (int tn = 0; tn < 2; ++tn) {
      const int col = bn * 128 + wave * 32 + tn * 16 + l15;
      const float bv = bias ? bias[col] : 0.f;
#pragma unroll
      for (int r = 0; r < 8; ++r) {
        const int row = bm * 64 + tm * 16 + r + half * 8;
        const size_t idx = (size_t)row * N + col;
        float v = acc[tm][tn][r] + bv;
        if (EPI == 1) {
          outB[idx] = (bf16_t)v;
        } else if (EPI == 2) {
          outB[idx] = (bf16_t)(0.5f * v * (1.f + erff(v * 0.70710678118f)));
        } else {
          outF[idx] = resid[idx] + v;
        }
      }
    }
  }
}

// ---------- Flash attention (bf16 in/out, f32 online softmax) ----------
// qkv: [4096, 2304] bf16 (q|k|v each 768 wide); vT: [48][64][1024] bf16.
// o: [4096, 768] bf16. grid: (16 q-blocks, 12 heads, 4 batches),
// 128 threads = 4 waves; each wave owns 16 query rows x 64-wide key chunk.
// K/V chunks are double-buffered through the async global->LDS path.
__global__ __launch_bounds__(128) void attention_kernel(
    const bf16_t* __restrict__ qkv, const bf16_t* __restrict__ vT,
    bf16_t* __restrict__ o) {
  __shared__ bf16_t Qs[64][64];
  __shared__ bf16_t Ks[2][64][64];
  __shared__ bf16_t Vts[2][64][64];   // Vts[buf][d][j]
  __shared__ bf16_t Pws[4][16][64];   // per-wave P tile

  const int tid = threadIdx.x;
  const int lane = tid & 31, wave = tid >> 5;
  const int half = lane >> 4, l15 = lane & 15;
  const int qb = blockIdx.x, h = blockIdx.y, b = blockIdx.z;
  const int bh = b * HEADS + h;

  // Stage Q block (64 x 64) via async copies
#pragma unroll
  for (int i = 0; i < 4; ++i) {
    int u = tid + i * 128;
    int r = u >> 3, c8 = (u & 7) * 8;
    int t = b * SEQ + qb * 64 + r;
    cp16(qkv + (size_t)t * QKVW + h * HDIM + c8, &Qs[r][c8]);
  }

  auto stageKV = [&](int buf, int j0) {
#pragma unroll
    for (int i = 0; i < 4; ++i) {
      int u = tid + i * 128;
      int r = u >> 3, c8 = (u & 7) * 8;   // K: r = key row, c8 = channel
      cp16(qkv + (size_t)(b * SEQ + j0 + r) * QKVW + INNERX + h * HDIM + c8,
           &Ks[buf][r][c8]);
      // V (pre-transposed): r = channel d, c8 = key offset
      cp16(vT + ((size_t)bh * HDIM + r) * SEQ + j0 + c8, &Vts[buf][r][c8]);
    }
  };

  float mprev[8], lsum[8];
  v8f oacc[4];
#pragma unroll
  for (int r = 0; r < 8; ++r) { mprev[r] = -1e30f; lsum[r] = 0.f; }
#pragma unroll
  for (int dt = 0; dt < 4; ++dt) oacc[dt] = zero8();

  stageKV(0, 0);
  for (int jc = 0; jc < SEQ / 64; ++jc) {
    const int buf = jc & 1;
    if (jc + 1 < SEQ / 64) {
      stageKV(buf ^ 1, (jc + 1) * 64);  // prefetch next K/V chunk
      WAIT_ASYNC(8);                    // current chunk's 8 copies done
    } else {
      WAIT_ASYNC(0);
    }
    __syncthreads();

    // S = Q Kᵀ for this wave's 16 rows (4 tiles of 16x16, K=64 in 2 steps)
    v8f s[4];
#pragma unroll
    for (int tn = 0; tn < 4; ++tn) {
      s[tn] = zero8();
#pragma unroll
      for (int kk = 0; kk < 2; ++kk) {
        v16bf af = load_frag(&Qs[wave * 16 + l15][kk * 32], half);
        v16bf bf_ = load_frag(&Ks[buf][tn * 16 + l15][kk * 32], half);
        s[tn] = wmma_bf16(af, bf_, s[tn]);
      }
    }
#pragma unroll
    for (int tn = 0; tn < 4; ++tn)
#pragma unroll
      for (int r = 0; r < 8; ++r) s[tn][r] *= SCALEF;

    // Online softmax: row stats live across 16 lanes of each half.
    float mx[8];
#pragma unroll
    for (int r = 0; r < 8; ++r)
      mx[r] = fmaxf(fmaxf(s[0][r], s[1][r]), fmaxf(s[2][r], s[3][r]));
#pragma unroll
    for (int off = 8; off >= 1; off >>= 1)
#pragma unroll
      for (int r = 0; r < 8; ++r) mx[r] = fmaxf(mx[r], __shfl_xor(mx[r], off, 32));

    float sc[8], rs[8];
#pragma unroll
    for (int r = 0; r < 8; ++r) {
      float mnew = fmaxf(mprev[r], mx[r]);
      sc[r] = __expf(mprev[r] - mnew);
      mprev[r] = mnew;
      rs[r] = 0.f;
    }
#pragma unroll
    for (int tn = 0; tn < 4; ++tn)
#pragma unroll
      for (int r = 0; r < 8; ++r) {
        s[tn][r] = __expf(s[tn][r] - mprev[r]);
        rs[r] += s[tn][r];
      }
#pragma unroll
    for (int off = 8; off >= 1; off >>= 1)
#pragma unroll
      for (int r = 0; r < 8; ++r) rs[r] += __shfl_xor(rs[r], off, 32);
#pragma unroll
    for (int r = 0; r < 8; ++r) lsum[r] = lsum[r] * sc[r] + rs[r];
#pragma unroll
    for (int dt = 0; dt < 4; ++dt)
#pragma unroll
      for (int r = 0; r < 8; ++r) oacc[dt][r] *= sc[r];

    // D-layout -> A-layout via per-wave LDS round trip
#pragma unroll
    for (int tn = 0; tn < 4; ++tn)
#pragma unroll
      for (int r = 0; r < 8; ++r)
        Pws[wave][r + half * 8][tn * 16 + l15] = (bf16_t)s[tn][r];

    // O += P * V  (K=64 keys in 2 steps of 32)
#pragma unroll
    for (int dt = 0; dt < 4; ++dt)
#pragma unroll
      for (int kk = 0; kk < 2; ++kk) {
        v16bf af = load_frag(&Pws[wave][l15][kk * 32], half);
        v16bf bf_ = load_frag(&Vts[buf][dt * 16 + l15][kk * 32], half);
        oacc[dt] = wmma_bf16(af, bf_, oacc[dt]);
      }
    __syncthreads();
  }

  // Normalize and write out (bf16)
#pragma unroll
  for (int dt = 0; dt < 4; ++dt)
#pragma unroll
    for (int r = 0; r < 8; ++r) {
      int rl = wave * 16 + r + half * 8;
      int t = b * SEQ + qb * 64 + rl;
      int col = h * HDIM + dt * 16 + l15;
      o[(size_t)t * INNERX + col] = (bf16_t)(oacc[dt][r] / lsum[r]);
    }
}

// ---------- host orchestration ----------
extern "C" void kernel_launch(void* const* d_in, const int* in_sizes, int n_in,
                              void* d_out, int out_size, void* d_ws, size_t ws_size,
                              hipStream_t stream) {
  (void)in_sizes; (void)n_in; (void)out_size; (void)ws_size;
  const float* x    = (const float*)d_in[0];
  const float* Wqkv = (const float*)d_in[1];
  const float* Wout = (const float*)d_in[2];
  const float* bout = (const float*)d_in[3];
  const float* ln1g = (const float*)d_in[4];
  const float* ln1b = (const float*)d_in[5];
  const float* W1   = (const float*)d_in[6];
  const float* b1   = (const float*)d_in[7];
  const float* W2   = (const float*)d_in[8];
  const float* b2   = (const float*)d_in[9];
  const float* ln2g = (const float*)d_in[10];
  const float* ln2b = (const float*)d_in[11];
  const float* lnfg = (const float*)d_in[12];
  const float* lnfb = (const float*)d_in[13];
  float* h = (float*)d_out;   // residual stream lives in d_out (f32)

  char* ws = (char*)d_ws;
  size_t off = 0;
  auto alloc = [&](size_t bytes) -> char* {
    char* p = ws + off;
    off += (bytes + 255) & ~(size_t)255;
    return p;
  };
  // Pre-transposed bf16 weights: Wt[n][k] so GEMM B-tiles stage contiguously.
  bf16_t* WqkvT = (bf16_t*)alloc((size_t)NLAYER * DIMX * QKVW * sizeof(bf16_t));
  bf16_t* WoutT = (bf16_t*)alloc((size_t)NLAYER * INNERX * DIMX * sizeof(bf16_t));
  bf16_t* W1T   = (bf16_t*)alloc((size_t)NLAYER * DIMX * DFFX * sizeof(bf16_t));
  bf16_t* W2T   = (bf16_t*)alloc((size_t)NLAYER * DFFX * DIMX * sizeof(bf16_t));
  bf16_t* xln_b = (bf16_t*)alloc((size_t)ROWS * DIMX * sizeof(bf16_t));
  bf16_t* qkv_b = (bf16_t*)alloc((size_t)ROWS * QKVW * sizeof(bf16_t));
  bf16_t* vT_b  = (bf16_t*)alloc((size_t)BATCH * HEADS * HDIM * SEQ * sizeof(bf16_t));
  bf16_t* o_b   = (bf16_t*)alloc((size_t)ROWS * INNERX * sizeof(bf16_t));
  bf16_t* hff_b = (bf16_t*)alloc((size_t)ROWS * DFFX * sizeof(bf16_t));

  // Cast + transpose weights once per launch (deterministic).
  cvtT_kernel<<<dim3(QKVW / 32, DIMX / 32, NLAYER), dim3(32, 8), 0, stream>>>(
      Wqkv, WqkvT, DIMX, QKVW);
  cvtT_kernel<<<dim3(DIMX / 32, INNERX / 32, NLAYER), dim3(32, 8), 0, stream>>>(
      Wout, WoutT, INNERX, DIMX);
  cvtT_kernel<<<dim3(DFFX / 32, DIMX / 32, NLAYER), dim3(32, 8), 0, stream>>>(
      W1, W1T, DIMX, DFFX);
  cvtT_kernel<<<dim3(DIMX / 32, DFFX / 32, NLAYER), dim3(32, 8), 0, stream>>>(
      W2, W2T, DFFX, DIMX);

  {
    int n = ROWS * DIMX;
    copy_f32_kernel<<<(n + 255) / 256, 256, 0, stream>>>(x, h, n);
  }

  for (int l = 0; l < NLAYER; ++l) {
    // ---- attention block ----
    layernorm_kernel<true><<<ROWS, 256, 0, stream>>>(
        h, ln1g + l * DIMX, ln1b + l * DIMX, xln_b, nullptr);
    gemm_bf16_kernel<1><<<dim3(QKVW / 128, ROWS / 64), 128, 0, stream>>>(
        xln_b, WqkvT + (size_t)l * DIMX * QKVW, nullptr, nullptr, nullptr,
        qkv_b, ROWS, QKVW, DIMX);
    transposeV_kernel<<<dim3(SEQ / 32, HDIM / 32, BATCH * HEADS), dim3(32, 8),
                        0, stream>>>(qkv_b, vT_b);
    attention_kernel<<<dim3(SEQ / 64, HEADS, BATCH), 128, 0, stream>>>(
        qkv_b, vT_b, o_b);
    gemm_bf16_kernel<3><<<dim3(DIMX / 128, ROWS / 64), 128, 0, stream>>>(
        o_b, WoutT + (size_t)l * INNERX * DIMX, bout + l * DIMX, h, h, nullptr,
        ROWS, DIMX, INNERX);
    // ---- MLP block ----
    layernorm_kernel<true><<<ROWS, 256, 0, stream>>>(
        h, ln2g + l * DIMX, ln2b + l * DIMX, xln_b, nullptr);
    gemm_bf16_kernel<2><<<dim3(DFFX / 128, ROWS / 64), 128, 0, stream>>>(
        xln_b, W1T + (size_t)l * DIMX * DFFX, b1 + l * DFFX, nullptr, nullptr,
        hff_b, ROWS, DFFX, DIMX);
    gemm_bf16_kernel<3><<<dim3(DIMX / 128, ROWS / 64), 128, 0, stream>>>(
        hff_b, W2T + (size_t)l * DFFX * DIMX, b2 + l * DIMX, h, h, nullptr,
        ROWS, DIMX, DFFX);
  }

  // Final LayerNorm, f32 in-place on d_out
  layernorm_kernel<false><<<ROWS, 256, 0, stream>>>(h, lnfg, lnfb, nullptr, h);
}
